// NeuralSynth_49941879718232
// MI455X (gfx1250) — compile-verified, matrix operand order checked by reading, not run on hardware
//
#include <hip/hip_runtime.h>

// ---------------------------------------------------------------------------
// DDSP-style neural synth for MI455X (gfx1250, wave32, WMMA).
// GEMMs / GRU recurrence / reverb convolution use v_wmma_f32_16x16x32_f16.
// k_noise stages its input chunk through the Tensor Data Mover (TDM).
// ---------------------------------------------------------------------------

typedef __attribute__((ext_vector_type(16))) _Float16 v16h;
typedef __attribute__((ext_vector_type(8)))  _Float16 v8h;
typedef __attribute__((ext_vector_type(8)))  float    v8f;
typedef unsigned int u32x4 __attribute__((ext_vector_type(4)));
typedef int          i32x8 __attribute__((ext_vector_type(8)));
typedef int          i32x4 __attribute__((ext_vector_type(4)));

#define NB_     8
#define SEQ_    400
#define HID_    512
#define P_      100
#define FS_     160
#define T_      64000
#define R_      3200      // B*SEQ
#define G3_     1536      // 3*HID

__device__ inline v8f wmma16(v16h a, v16h b, v8f c) {
  return __builtin_amdgcn_wmma_f32_16x16x32_f16(false, a, false, b, (short)0, c,
                                                false, false);
}

// Load a 16-f16 fragment: 8 halves at p, 8 halves at p+16 (layout: K and K+16)
__device__ inline v16h load_frag(const _Float16* p) {
  v8h lo = *(const v8h*)(p);
  v8h hi = *(const v8h*)(p + 16);
  v16h r;
#pragma unroll
  for (int i = 0; i < 8; ++i) { r[i] = lo[i]; r[i + 8] = hi[i]; }
  return r;
}

// ------------------------------- utility -----------------------------------
__global__ void k_cvt(const float* __restrict__ s, _Float16* __restrict__ d, int n) {
  int i = blockIdx.x * blockDim.x + threadIdx.x;
  if (i < n) d[i] = (_Float16)s[i];
}

__global__ void k_zero32(unsigned int* p, int n) {
  int i = blockIdx.x * blockDim.x + threadIdx.x;
  if (i < n) p[i] = 0u;
}

// ------------------------- MLP scalar-input layer ---------------------------
__global__ void k_layer0(const float* __restrict__ x, const float* __restrict__ W,
                         const float* __restrict__ b, float* __restrict__ pre) {
  int i = blockIdx.x * blockDim.x + threadIdx.x;
  if (i >= R_ * HID_) return;
  int r = i >> 9, j = i & 511;
  pre[i] = x[r] * W[j] + b[j];
}

// ----------------------- LayerNorm + ReLU -> f16 act ------------------------
__global__ __launch_bounds__(256)
void k_lnrelu(const float* __restrict__ pre, const float* __restrict__ g,
              const float* __restrict__ beta, _Float16* __restrict__ outh,
              int ostride, int ocol) {
  __shared__ float s1[256], s2[256];
  int r = blockIdx.x, tid = threadIdx.x;
  float a = pre[(size_t)r * HID_ + tid];
  float b = pre[(size_t)r * HID_ + tid + 256];
  s1[tid] = a + b;
  s2[tid] = a * a + b * b;
  __syncthreads();
  for (int off = 128; off > 0; off >>= 1) {
    if (tid < off) { s1[tid] += s1[tid + off]; s2[tid] += s2[tid + off]; }
    __syncthreads();
  }
  float mu  = s1[0] * (1.0f / 512.0f);
  float var = s2[0] * (1.0f / 512.0f) - mu * mu;
  float rs  = rsqrtf(var + 1e-5f);
  float o1 = fmaxf(0.0f, (a - mu) * rs * g[tid] + beta[tid]);
  float o2 = fmaxf(0.0f, (b - mu) * rs * g[tid + 256] + beta[tid + 256]);
  outh[(size_t)r * ostride + ocol + tid]       = (_Float16)o1;
  outh[(size_t)r * ostride + ocol + tid + 256] = (_Float16)o2;
}

// ------------------------------ WMMA GEMM -----------------------------------
// C[M,N](f32) = A[M,K](f16, row-major) * B[N,K](f16, row-major)^T + bias[N]
__global__ void k_gemm(const _Float16* __restrict__ A, const _Float16* __restrict__ Bm,
                       const float* __restrict__ bias, float* __restrict__ C,
                       int M, int N, int K) {
  int wid  = blockIdx.x * (blockDim.x >> 5) + (threadIdx.x >> 5);
  int nt_n = N >> 4;
  if (wid >= (M >> 4) * nt_n) return;
  int mt = wid / nt_n, nt = wid - mt * nt_n;
  int lane = threadIdx.x & 31, sel = lane >> 4, l15 = lane & 15;
  const _Float16* Ap = A  + (size_t)(mt * 16 + l15) * K + sel * 8;
  const _Float16* Bp = Bm + (size_t)(nt * 16 + l15) * K + sel * 8;
  v8f acc = {};
  for (int k0 = 0; k0 < K; k0 += 32) {
    __builtin_prefetch(Ap + k0 + 128, 0, 0);
    __builtin_prefetch(Bp + k0 + 128, 0, 0);
    acc = wmma16(load_frag(Ap + k0), load_frag(Bp + k0), acc);
  }
  int n = nt * 16 + l15;
  float bv = bias ? bias[n] : 0.0f;
#pragma unroll
  for (int v = 0; v < 8; ++v)
    C[(size_t)(mt * 16 + sel * 8 + v) * N + n] = acc[v] + bv;
}

// -------------------------- persistent GRU kernel ---------------------------
// One workgroup, 32 waves. Hidden state in registers; f16 mirror in LDS feeds
// WMMA A-fragments (ds_load_b128); Wh streamed from global as B-fragments.
__global__ __launch_bounds__(1024)
void k_gru(const float* __restrict__ gi, const _Float16* __restrict__ Wh,
           const float* __restrict__ bh, _Float16* __restrict__ ys) {
  __shared__ _Float16 hsh[16][520];   // rows 8..15 stay zero (M padding)
  __shared__ _Float16 gh[8][1552];    // staged gates (r|z|n), padded stride
  int tid = threadIdx.x;
  int lane = tid & 31, wave = tid >> 5;
  int sel = lane >> 4, l15 = lane & 15;

  for (int i = tid; i < 16 * 520; i += 1024) (&hsh[0][0])[i] = (_Float16)0.0f;
  float hreg[4] = {0.0f, 0.0f, 0.0f, 0.0f};
  __syncthreads();

  const _Float16* b0 = Wh + (size_t)((wave +  0) * 16 + l15) * HID_ + sel * 8;
  const _Float16* b1 = Wh + (size_t)((wave + 32) * 16 + l15) * HID_ + sel * 8;
  const _Float16* b2 = Wh + (size_t)((wave + 64) * 16 + l15) * HID_ + sel * 8;

  for (int t = 0; t < SEQ_; ++t) {
    // publish h -> LDS (f16)
#pragma unroll
    for (int u = 0; u < 4; ++u) {
      int f = tid * 4 + u;
      hsh[f >> 9][f & 511] = (_Float16)hreg[u];
    }
    __syncthreads();

    // gh = h @ Wh^T : each wave owns N-tiles {wave, wave+32, wave+64}
    v8f a0 = {}, a1 = {}, a2 = {};
#pragma unroll 4
    for (int k0 = 0; k0 < HID_; k0 += 32) {
      v16h af = load_frag(&hsh[l15][k0 + sel * 8]);
      a0 = wmma16(af, load_frag(b0 + k0), a0);
      a1 = wmma16(af, load_frag(b1 + k0), a1);
      a2 = wmma16(af, load_frag(b2 + k0), a2);
    }
    if (sel == 0) {   // lanes 0-15 hold batch rows 0-7 of D
      int c0 = (wave +  0) * 16 + l15;
      int c1 = (wave + 32) * 16 + l15;
      int c2 = (wave + 64) * 16 + l15;
#pragma unroll
      for (int v = 0; v < 8; ++v) {
        gh[v][c0] = (_Float16)(a0[v] + bh[c0]);
        gh[v][c1] = (_Float16)(a1[v] + bh[c1]);
        gh[v][c2] = (_Float16)(a2[v] + bh[c2]);
      }
    }
    __syncthreads();

    // gate math + state update
#pragma unroll
    for (int u = 0; u < 4; ++u) {
      int f = tid * 4 + u;
      int b = f >> 9, j = f & 511;
      const float* gp = gi + ((size_t)(b * SEQ_ + t)) * G3_ + j;
      float ir = gp[0], iz = gp[512], in = gp[1024];
      float hr = (float)gh[b][j], hz = (float)gh[b][j + 512], hn = (float)gh[b][j + 1024];
      float r = 1.0f / (1.0f + __expf(-(ir + hr)));
      float z = 1.0f / (1.0f + __expf(-(iz + hz)));
      float n = tanhf(in + r * hn);
      float hp = (1.0f - z) * n + z * hreg[u];
      hreg[u] = hp;
      ys[((size_t)(b * SEQ_ + t)) * HID_ + j] = (_Float16)hp;
    }
    __syncthreads();
  }
}

// -------------------------- projection heads --------------------------------
__global__ void k_head(const _Float16* __restrict__ act, const float* __restrict__ W,
                       const float* __restrict__ b, float* __restrict__ out, int nout) {
  int i = blockIdx.x * blockDim.x + threadIdx.x;
  if (i >= R_ * nout) return;
  int r = i / nout, j = i - r * nout;
  const _Float16* a = act + (size_t)r * HID_;
  const float*    w = W   + (size_t)j * HID_;
  float s = 0.0f;
#pragma unroll 8
  for (int k = 0; k < HID_; ++k) s += (float)a[k] * w[k];
  s += b[j];
  float sg = 1.0f / (1.0f + __expf(-s));
  out[i] = 2.0f * __powf(sg, 2.302585093f) + 1e-7f;     // mod_sigmoid
}

__global__ void k_alpha_norm(float* alpha) {
  int r = blockIdx.x * blockDim.x + threadIdx.x;
  if (r >= R_) return;
  float s = 0.0f;
  for (int p = 0; p < P_; ++p) s += alpha[r * P_ + p];
  float inv = 1.0f / s;
  for (int p = 0; p < P_; ++p) alpha[r * P_ + p] *= inv;
}

// -------------------- upsampled f0, phase scan, synth -----------------------
__global__ void k_f0u(const float* __restrict__ f0, float* __restrict__ f0u) {
  int i = blockIdx.x * blockDim.x + threadIdx.x;
  if (i >= NB_ * T_) return;
  int b = i / T_, t = i - b * T_;
  float src = fmaxf((t + 0.5f) * (1.0f / 160.0f) - 0.5f, 0.0f);
  int i0 = (int)src;
  int i1 = min(i0 + 1, SEQ_ - 1);
  float w = src - (float)i0;
  float v = f0[b * SEQ_ + i0] * (1.0f - w) + f0[b * SEQ_ + i1] * w;
  f0u[i] = v * (1.0f / 16000.0f);
}

__global__ __launch_bounds__(1024)
void k_scan(const float* __restrict__ f0u, float* __restrict__ phi) {
  __shared__ float sb[1024];
  __shared__ float carry;
  int b = blockIdx.x, tid = threadIdx.x;
  if (tid == 0) carry = 0.0f;
  __syncthreads();
  for (int c = 0; c < 63; ++c) {
    int t = c * 1024 + tid;
    float v = (t > 0 && t < T_) ? 6.28318530717958647f * f0u[b * T_ + t] : 0.0f;
    sb[tid] = v;
    __syncthreads();
    for (int off = 1; off < 1024; off <<= 1) {
      float x = (tid >= off) ? sb[tid - off] : 0.0f;
      __syncthreads();
      sb[tid] += x;
      __syncthreads();
    }
    if (t < T_) phi[b * T_ + t] = sb[tid] + carry;
    __syncthreads();
    if (tid == 1023) carry += sb[1023];
    __syncthreads();
  }
}

__global__ void k_synth(const float* __restrict__ ampb, const float* __restrict__ alphab,
                        const float* __restrict__ f0u, const float* __restrict__ phi,
                        float* __restrict__ out_ampu, float* __restrict__ out_alphau,
                        _Float16* __restrict__ yf) {
  int i = blockIdx.x * blockDim.x + threadIdx.x;
  if (i >= NB_ * T_) return;
  int b = i / T_, t = i - b * T_;
  float src = fmaxf((t + 0.5f) * (1.0f / 160.0f) - 0.5f, 0.0f);
  int i0 = (int)src;
  int i1 = min(i0 + 1, SEQ_ - 1);
  float w = src - (float)i0, w1 = 1.0f - w;
  int r0 = b * SEQ_ + i0, r1 = b * SEQ_ + i1;
  float ampu = ampb[r0] * w1 + ampb[r1] * w;
  out_ampu[i] = ampu;
  float f = f0u[i], ph = phi[i];
  const float* a0 = alphab + (size_t)r0 * P_;
  const float* a1 = alphab + (size_t)r1 * P_;
  float* oa = out_alphau + (size_t)i * P_;
  float s = 0.0f;
#pragma unroll 4
  for (int p = 0; p < P_; ++p) {
    float al = a0[p] * w1 + a1[p] * w;
    oa[p] = al;
    float k = (float)(p + 1);
    if (k * f < 0.5f) s += al * __sinf(k * ph);
  }
  yf[(size_t)b * T_ + t] = (_Float16)(ampu * s);
}

// ------------------------------ reverb IR -----------------------------------
__global__ void k_impulse(const float* __restrict__ decay, const float* __restrict__ wetdry,
                          const float* __restrict__ ident, const float* __restrict__ imp,
                          _Float16* __restrict__ himp) {
  int t = blockIdx.x * blockDim.x + threadIdx.x;
  if (t >= T_) return;
  float wd  = wetdry[0];
  float sw  = 1.0f / (1.0f + __expf(-wd));
  float sw2 = 1.0f / (1.0f + __expf(-(1.0f - wd)));
  float dcy = __expf(-(__expf(decay[0]) + 2.0f) * ((float)t * (1.0f / 63999.0f)));
  float v = sw * ident[t] + sw2 * imp[t] * dcy;
  himp[t] = (_Float16)tanhf(v);
}

// ------------------ Toeplitz-tiled WMMA causal convolution ------------------
// yout[b][t0+i] = sum_tau h[t0+i-tau] * y[b][tau]
// A[i][k] = h[(t0-tau0) + i - k] (per-lane gather), B[k][b] = y[b][tau0+k].
__global__ void k_conv(const _Float16* __restrict__ yf, const _Float16* __restrict__ himp,
                       float* __restrict__ yout) {
  int wid = blockIdx.x * (blockDim.x >> 5) + (threadIdx.x >> 5);
  if (wid >= T_ / 16) return;
  int t0 = wid << 4;
  int lane = threadIdx.x & 31, sel = lane >> 4, l15 = lane & 15;
  const _Float16* yrow = yf + (size_t)l15 * T_ + sel * 8;   // N = batch slot
  v8f acc = {};
  for (int tau0 = 0; tau0 <= t0; tau0 += 32) {
    __builtin_prefetch(yrow + tau0 + 64, 0, 0);
    v16h bf = load_frag(yrow + tau0);
    int delta = t0 - tau0 + l15 - sel * 8;   // M = l15
    v16h af;
#pragma unroll
    for (int e = 0; e < 8; ++e) {
      int h0 = delta - e;
      int h1 = delta - 16 - e;
      af[e]     = (h0 >= 0) ? himp[h0] : (_Float16)0.0f;
      af[e + 8] = (h1 >= 0) ? himp[h1] : (_Float16)0.0f;
    }
    acc = wmma16(af, bf, acc);
  }
  if (l15 < NB_) {
#pragma unroll
    for (int v = 0; v < 8; ++v)
      yout[(size_t)l15 * T_ + t0 + sel * 8 + v] = acc[v];
  }
}

// ------------------- noise shaping: direct 160-pt DFTs ----------------------
// Input chunk is staged LDS-side through the Tensor Data Mover when available.
__global__ __launch_bounds__(160)
void k_noise(const float* __restrict__ noise, const float* __restrict__ filt,
             float* __restrict__ out4) {
  __shared__ float g[FS_], nz[FS_];
  int c = blockIdx.x, n = threadIdx.x;
  const float* fc = filt + (size_t)c * 81;
#if __has_builtin(__builtin_amdgcn_tensor_load_to_lds) && \
    __has_builtin(__builtin_amdgcn_s_wait_tensorcnt)
  if (threadIdx.x < 32) {   // wave 0 issues one TDM descriptor (EXEC ignored)
    unsigned long long ga = (unsigned long long)(size_t)(noise + (size_t)c * FS_);
    unsigned lds = (unsigned)(size_t)(void*)nz;   // generic low 32 = LDS offset
    u32x4 g0;  i32x8 g1;  i32x4 gz;  i32x8 gz8;
    g0[0] = 1u;                                         // count=1 valid descriptor
    g0[1] = lds;                                        // lds_addr (bytes)
    g0[2] = (unsigned)(ga & 0xFFFFFFFFull);             // global_addr[31:0]
    g0[3] = (unsigned)((ga >> 32) & 0x01FFFFFFull)      // global_addr[56:32]
          | 0x80000000u;                                // type = 2 ("image")
    g1[0] = (int)(2u << 16);                            // data_size = 4 bytes
    g1[1] = (int)((FS_ & 0xFFFF) << 16);                // tensor_dim0[15:0]
    g1[2] = (int)((FS_ >> 16) | (1u << 16));            // dim0[31:16] | tensor_dim1=1
    g1[3] = (int)(FS_ << 16);                           // tile_dim0 = 160
    g1[4] = 0;                                          // tile_dim1/2 unused
    g1[5] = FS_;                                        // tensor_dim0_stride
    g1[6] = 0;  g1[7] = 0;
    gz[0] = 0; gz[1] = 0; gz[2] = 0; gz[3] = 0;         // <=2D: groups 2/3 off
    gz8[0] = 0; gz8[1] = 0; gz8[2] = 0; gz8[3] = 0;
    gz8[4] = 0; gz8[5] = 0; gz8[6] = 0; gz8[7] = 0;
    // 6-arg form (clang-23 / therock-10.0 lane, per compile diagnostic)
    __builtin_amdgcn_tensor_load_to_lds(g0, g1, gz, gz, gz8, 0);
    __builtin_amdgcn_s_wait_tensorcnt(0);
  }
  __syncthreads();
#else
  nz[n] = noise[(size_t)c * FS_ + n];
#endif
  const float w0 = 6.28318530717958647f / 160.0f;
  // h = irfft(fc) ; real symmetric
  float h = fc[0];
  for (int k = 1; k < 80; ++k) h += 2.0f * fc[k] * __cosf(w0 * (float)(k * n));
  h += fc[80] * ((n & 1) ? -1.0f : 1.0f);
  h *= (1.0f / 160.0f);
  float fwin = 0.5f * (1.0f + __cosf(w0 * (float)n));  // roll(hann, 80)
  g[n] = fwin * h;
  __syncthreads();
  if (n < 81) {
    float Hr = 0, Hi = 0, Nr = 0, Ni = 0;
    for (int m = 0; m < FS_; ++m) {
      float sn, cs;
      __sincosf(w0 * (float)(n * m), &sn, &cs);
      Hr += g[m] * cs;  Hi -= g[m] * sn;
      Nr += nz[m] * cs; Ni -= nz[m] * sn;
    }
    out4[((size_t)c * 81 + n) * 2 + 0] = Hr * Nr - Hi * Ni;
    out4[((size_t)c * 81 + n) * 2 + 1] = Hr * Ni + Hi * Nr;
  }
}

// ---------------------------------------------------------------------------
// Host driver
// ---------------------------------------------------------------------------
// d_in flattening assumption (JAX sorted-pytree leaf order):
//  0 f0  1 lo  2 noise
//  3 alpha.W  4 alpha.b  5 amp.W  6 amp.b
//  7..18  f0_mlp L0..L2 (W,b,beta,g each)
// 19..30  fi_mlp L0..L2 (W,b,beta,g each)
// 31 filt.W 32 filt.b
// 33 gru.Wh 34 gru.Wi 35 gru.bh 36 gru.bi
// 37..48  lo_mlp L0..L2 (W,b,beta,g each)
// 49 decay 50 identity 51 impulse 52 wetdry

static inline int gemm_blocks(int M, int N) { return ((M / 16) * (N / 16) + 7) / 8; }

extern "C" void kernel_launch(void* const* d_in, const int* in_sizes, int n_in,
                              void* d_out, int out_size, void* d_ws, size_t ws_size,
                              hipStream_t stream) {
  (void)in_sizes; (void)n_in; (void)out_size; (void)ws_size;
  const float* F0   = (const float*)d_in[0];
  const float* LO   = (const float*)d_in[1];
  const float* NOI  = (const float*)d_in[2];
  const float* aW   = (const float*)d_in[3];
  const float* ab   = (const float*)d_in[4];
  const float* mW   = (const float*)d_in[5];
  const float* mb   = (const float*)d_in[6];
  const float* fW   = (const float*)d_in[31];
  const float* fb   = (const float*)d_in[32];
  const float* Wh   = (const float*)d_in[33];
  const float* Wi   = (const float*)d_in[34];
  const float* bh   = (const float*)d_in[35];
  const float* bi   = (const float*)d_in[36];
  const float* dcay = (const float*)d_in[49];
  const float* idnt = (const float*)d_in[50];
  const float* impl = (const float*)d_in[51];
  const float* wdry = (const float*)d_in[52];

  // ---- f16 weight pool offsets (elements) ----
  const size_t WF0_1 = 0,        WF0_2 = 262144;
  const size_t WLO_1 = 524288,   WLO_2 = 786432;
  const size_t WFI_0 = 1048576,  WFI_1 = 1310720, WFI_2 = 1572864;
  const size_t WWI   = 1835008;  // 1536*1024
  const size_t WWH   = 3407872;  // 1536*512
  // ---- workspace layout (bytes) ----
  char* ws = (char*)d_ws;
  _Float16* wf16  = (_Float16*)(ws + 0);                 //  8,388,608 B
  float*    pre   = (float*)   (ws + 8388608);           //  6,553,600 B
  float*    gi    = (float*)   (ws + 14942208);          // 19,660,800 B
  _Float16* xcat  = (_Float16*)(ws + 34603008);          //  6,553,600 B
  _Float16* atmp  = (_Float16*)(ws + 41156608);          //  3,276,800 B
  _Float16* ysh   = (_Float16*)(ws + 44433408);          //  3,276,800 B
  float*    ampb  = (float*)   (ws + 47710208);          //     12,800 B
  float*    alphb = (float*)   (ws + 47723008);          //  1,280,000 B
  float*    filtb = (float*)   (ws + 49003008);          //  1,036,800 B
  float*    f0u   = (float*)   (ws + 50039808);          //  2,048,000 B
  float*    phi   = (float*)   (ws + 52087808);          //  2,048,000 B
  _Float16* yf16  = (_Float16*)(ws + 54135808);          //  2,048,000 B (16 x T)
  _Float16* himp  = (_Float16*)(ws + 56183808);          //    128,000 B

  float* out1 = (float*)d_out;              // yout   [8,64000]
  float* out2 = out1 + 512000;              // ampu   [8,64000]
  float* out3 = out2 + 512000;              // alphau [8,64000,100]
  float* out4 = out3 + 51200000;            // S_filt_ri [8,400,81,2]

  auto cvt = [&](const float* s, size_t off, int n) {
    k_cvt<<<(n + 255) / 256, 256, 0, stream>>>(s, wf16 + off, n);
  };
  // weight conversion f32 -> f16
  cvt((const float*)d_in[11], WF0_1, 262144);
  cvt((const float*)d_in[15], WF0_2, 262144);
  cvt((const float*)d_in[41], WLO_1, 262144);
  cvt((const float*)d_in[45], WLO_2, 262144);
  cvt((const float*)d_in[19], WFI_0, 262144);
  cvt((const float*)d_in[23], WFI_1, 262144);
  cvt((const float*)d_in[27], WFI_2, 262144);
  cvt(Wi, WWI, 1572864);
  cvt(Wh, WWH, 786432);

  const int E = R_ * HID_;  // 1,638,400 elementwise MLP entries

  // ---- f0 MLP -> xcat[:, 0:512] ----
  k_layer0<<<(E + 255) / 256, 256, 0, stream>>>(F0, (const float*)d_in[7],
                                                (const float*)d_in[8], pre);
  k_lnrelu<<<R_, 256, 0, stream>>>(pre, (const float*)d_in[10], (const float*)d_in[9],
                                   atmp, HID_, 0);
  k_gemm<<<gemm_blocks(R_, HID_), 256, 0, stream>>>(atmp, wf16 + WF0_1,
      (const float*)d_in[12], pre, R_, HID_, HID_);
  k_lnrelu<<<R_, 256, 0, stream>>>(pre, (const float*)d_in[14], (const float*)d_in[13],
                                   atmp, HID_, 0);
  k_gemm<<<gemm_blocks(R_, HID_), 256, 0, stream>>>(atmp, wf16 + WF0_2,
      (const float*)d_in[16], pre, R_, HID_, HID_);
  k_lnrelu<<<R_, 256, 0, stream>>>(pre, (const float*)d_in[18], (const float*)d_in[17],
                                   xcat, 2 * HID_, 0);

  // ---- lo MLP -> xcat[:, 512:1024] ----
  k_layer0<<<(E + 255) / 256, 256, 0, stream>>>(LO, (const float*)d_in[37],
                                                (const float*)d_in[38], pre);
  k_lnrelu<<<R_, 256, 0, stream>>>(pre, (const float*)d_in[40], (const float*)d_in[39],
                                   atmp, HID_, 0);
  k_gemm<<<gemm_blocks(R_, HID_), 256, 0, stream>>>(atmp, wf16 + WLO_1,
      (const float*)d_in[42], pre, R_, HID_, HID_);
  k_lnrelu<<<R_, 256, 0, stream>>>(pre, (const float*)d_in[44], (const float*)d_in[43],
                                   atmp, HID_, 0);
  k_gemm<<<gemm_blocks(R_, HID_), 256, 0, stream>>>(atmp, wf16 + WLO_2,
      (const float*)d_in[46], pre, R_, HID_, HID_);
  k_lnrelu<<<R_, 256, 0, stream>>>(pre, (const float*)d_in[48], (const float*)d_in[47],
                                   xcat, 2 * HID_, HID_);

  // ---- GRU: gi = xcat @ Wi^T + bi, then sequential recurrence ----
  k_gemm<<<gemm_blocks(R_, G3_), 256, 0, stream>>>(xcat, wf16 + WWI, bi, gi,
                                                   R_, G3_, 2 * HID_);
  k_gru<<<1, 1024, 0, stream>>>(gi, wf16 + WWH, bh, ysh);

  // ---- fi MLP ----
  k_gemm<<<gemm_blocks(R_, HID_), 256, 0, stream>>>(ysh, wf16 + WFI_0,
      (const float*)d_in[20], pre, R_, HID_, HID_);
  k_lnrelu<<<R_, 256, 0, stream>>>(pre, (const float*)d_in[22], (const float*)d_in[21],
                                   atmp, HID_, 0);
  k_gemm<<<gemm_blocks(R_, HID_), 256, 0, stream>>>(atmp, wf16 + WFI_1,
      (const float*)d_in[24], pre, R_, HID_, HID_);
  k_lnrelu<<<R_, 256, 0, stream>>>(pre, (const float*)d_in[26], (const float*)d_in[25],
                                   atmp, HID_, 0);
  k_gemm<<<gemm_blocks(R_, HID_), 256, 0, stream>>>(atmp, wf16 + WFI_2,
      (const float*)d_in[28], pre, R_, HID_, HID_);
  k_lnrelu<<<R_, 256, 0, stream>>>(pre, (const float*)d_in[30], (const float*)d_in[29],
                                   atmp, HID_, 0);

  // ---- heads ----
  k_head<<<(R_ * 1 + 255) / 256, 256, 0, stream>>>(atmp, mW, mb, ampb, 1);
  k_head<<<(R_ * P_ + 255) / 256, 256, 0, stream>>>(atmp, aW, ab, alphb, P_);
  k_alpha_norm<<<(R_ + 255) / 256, 256, 0, stream>>>(alphb);
  k_head<<<(R_ * 81 + 255) / 256, 256, 0, stream>>>(atmp, fW, fb, filtb, 81);

  // ---- audio-rate synthesis ----
  k_f0u<<<(NB_ * T_ + 255) / 256, 256, 0, stream>>>(F0, f0u);
  k_scan<<<NB_, 1024, 0, stream>>>(f0u, phi);
  k_zero32<<<(512000 + 255) / 256, 256, 0, stream>>>((unsigned int*)yf16, 512000);
  k_synth<<<(NB_ * T_ + 255) / 256, 256, 0, stream>>>(ampb, alphb, f0u, phi,
                                                      out2, out3, yf16);

  // ---- noise shaping spectra -> out4 (TDM-staged input) ----
  k_noise<<<R_, 160, 0, stream>>>(NOI, filtb, out4);

  // ---- reverb: WMMA Toeplitz convolution -> out1 ----
  k_impulse<<<(T_ + 255) / 256, 256, 0, stream>>>(dcay, wdry, idnt, impl, himp);
  k_conv<<<(T_ / 16 + 7) / 8, 256, 0, stream>>>(yf16, himp, out1);
}
